// SelfAttention_46995532153004
// MI455X (gfx1250) — compile-verified, matrix operand order checked by reading, not run on hardware
//
#include <hip/hip_runtime.h>
#include <hip/hip_bf16.h>
#include <math.h>

// ---------------------------------------------------------------------------
// CDNA5 (gfx1250, wave32): yat-dense QKV + RoPE + causal flash attention +
// yat-dense proj. Matmuls: v_wmma_f32_16x16x32_bf16. Staging: double-buffered
// LDS filled by global_load_async_to_lds_b128 (ASYNCcnt path); the 16B step
// between consecutive fills is baked into the instruction offset (it applies
// to both the LDS and the global address per the ISA).
// ---------------------------------------------------------------------------

typedef __attribute__((ext_vector_type(16))) __bf16       v16bf;
typedef __attribute__((ext_vector_type(8)))  float        v8f;
typedef __attribute__((ext_vector_type(8)))  unsigned int v8u;

#define B_    2
#define T_    2048
#define C_    1024
#define H_    16
#define D_    64
#define M_    (B_ * T_)    // 4096 rows
#define NQKV_ (3 * C_)     // 3072
#define EPSY  1e-6f

// ---- CDNA5 async memory->LDS ----------------------------------------------
__device__ __forceinline__ unsigned lds_off(const void* p) {
  // local->generic casts put the LDS offset in the low 32 bits.
  return (unsigned)(unsigned long long)p;
}
// Copy 64B (this thread's share of a 128B tile row): 4 x b128, offsets baked
// into the instruction (applies to both LDS dst and global src).
__device__ __forceinline__ void async_cp64B(unsigned lds, const __bf16* g) {
  asm volatile("global_load_async_to_lds_b128 %0, %1, off"
               :: "v"(lds), "v"(g) : "memory");
  asm volatile("global_load_async_to_lds_b128 %0, %1, off offset:16"
               :: "v"(lds), "v"(g) : "memory");
  asm volatile("global_load_async_to_lds_b128 %0, %1, off offset:32"
               :: "v"(lds), "v"(g) : "memory");
  asm volatile("global_load_async_to_lds_b128 %0, %1, off offset:48"
               :: "v"(lds), "v"(g) : "memory");
}
__device__ __forceinline__ void wait_async0() {
  asm volatile("s_wait_asynccnt 0x0" ::: "memory");
}

// ---- WMMA helpers ---------------------------------------------------------
// Shared lane->k mapping for A (16x32, row per lane) and B (32x16, col per
// lane, k-pairs contiguous because the B source is stored [n][k] row-major).
__device__ __forceinline__ v16bf frag_ld(const __bf16* base, int row, int ldk, int lane) {
  int hi = (lane >> 4) & 1;
  const __bf16* rp = base + row * ldk;
  v8u t;
#pragma unroll
  for (int v = 0; v < 8; ++v) {
    int kp = ((v >> 2) << 3) + (hi << 2) + (v & 3);   // k = 2*kp
    t[v] = *(const unsigned int*)(rp + 2 * kp);
  }
  return __builtin_bit_cast(v16bf, t);
}

__device__ __forceinline__ v8f wmma_bf16(v16bf a, v16bf b, v8f c) {
  return __builtin_amdgcn_wmma_f32_16x16x32_bf16(false, a, false, b, (short)0, c,
                                                 false, false);
}

__device__ __forceinline__ v8f vzero8() {
  v8f z = {0.f, 0.f, 0.f, 0.f, 0.f, 0.f, 0.f, 0.f};
  return z;
}

// ---- prep kernels ---------------------------------------------------------
__global__ __launch_bounds__(256) void k_convert_rowsq(const float* __restrict__ in,
                                                       __bf16* __restrict__ out,
                                                       float* __restrict__ rsq) {
  __shared__ float red[256];
  int r = blockIdx.x, t = threadIdx.x;
  const float4 v = ((const float4*)(in + (size_t)r * C_))[t];
  __bf16* o = out + (size_t)r * C_ + t * 4;
  o[0] = (__bf16)v.x; o[1] = (__bf16)v.y; o[2] = (__bf16)v.z; o[3] = (__bf16)v.w;
  red[t] = v.x * v.x + v.y * v.y + v.z * v.z + v.w * v.w;
  __syncthreads();
  for (int s = 128; s > 0; s >>= 1) {
    if (t < s) red[t] += red[t + s];
    __syncthreads();
  }
  if (t == 0) rsq[r] = red[0];
}

__global__ __launch_bounds__(256) void k_rowsq_bf16(const __bf16* __restrict__ in,
                                                    float* __restrict__ rsq) {
  __shared__ float red[256];
  int r = blockIdx.x, t = threadIdx.x;
  const __bf16* p = in + (size_t)r * C_ + t * 4;
  float a = (float)p[0], b = (float)p[1], c = (float)p[2], d = (float)p[3];
  red[t] = a * a + b * b + c * c + d * d;
  __syncthreads();
  for (int s = 128; s > 0; s >>= 1) {
    if (t < s) red[t] += red[t + s];
    __syncthreads();
  }
  if (t == 0) rsq[r] = red[0];
}

// fp32 [K][N] -> bf16 [N][K]
__global__ __launch_bounds__(256) void k_transpose_bf16(const float* __restrict__ in,
                                                        __bf16* __restrict__ out,
                                                        int K, int N) {
  __shared__ float tile[32][33];
  int tx = threadIdx.x & 31, ty = threadIdx.x >> 5;
  int n0 = blockIdx.x * 32, k0 = blockIdx.y * 32;
#pragma unroll
  for (int i = 0; i < 4; ++i) {
    int r = ty + i * 8;
    tile[r][tx] = in[(size_t)(k0 + r) * N + (n0 + tx)];
  }
  __syncthreads();
#pragma unroll
  for (int i = 0; i < 4; ++i) {
    int r = ty + i * 8;
    out[(size_t)(n0 + r) * K + (k0 + tx)] = (__bf16)tile[tx][r];
  }
}

// ---- yat-dense GEMM (MODE 0: -> q/k with RoPE + vT; MODE 1: -> fp32 out) --
// 128x128 block tile, 8 waves (4x2), double-buffered async LDS staging.
template <int MODE>
__global__ __launch_bounds__(256) void k_yat_gemm(
    const __bf16* __restrict__ A, const __bf16* __restrict__ BT,
    const float* __restrict__ a_sq, const float* __restrict__ b_sq,
    const float* __restrict__ bias, const float* __restrict__ alpha,
    int N, int outF, float* __restrict__ out32, __bf16* __restrict__ qh,
    __bf16* __restrict__ kh, __bf16* __restrict__ vT) {
  constexpr int K = C_;
  __shared__ __bf16 As[2][128 * 72];  // 128 rows x 64 k, stride 72 (pad)
  __shared__ __bf16 Bs[2][128 * 72];
  int t = threadIdx.x, lane = t & 31, wave = t >> 5;
  int wm = wave >> 1, wn = wave & 1;      // 4x2 wave grid, wave = 32x64
  int m0 = blockIdx.y * 128, n0 = blockIdx.x * 128;
  int srow = t >> 1, spart = t & 1;       // staging: 2 threads per row (64B each)
  int ln = lane & 15, hi = (lane >> 4) & 1;

  const __bf16* gA = A + (size_t)(m0 + srow) * K + spart * 32;
  const __bf16* gB = BT + (size_t)(n0 + srow) * K + spart * 32;
  unsigned la[2], lb[2];
  la[0] = lds_off(&As[0][srow * 72 + spart * 32]);
  la[1] = lds_off(&As[1][srow * 72 + spart * 32]);
  lb[0] = lds_off(&Bs[0][srow * 72 + spart * 32]);
  lb[1] = lds_off(&Bs[1][srow * 72 + spart * 32]);

  // prefetch chunk 0 into buffer 0
  async_cp64B(la[0], gA);
  async_cp64B(lb[0], gB);

  v8f acc[2][4];
#pragma unroll
  for (int mi = 0; mi < 2; ++mi)
#pragma unroll
    for (int ni = 0; ni < 4; ++ni) acc[mi][ni] = vzero8();

  for (int kc = 0; kc < K; kc += 64) {
    int buf = (kc >> 6) & 1;
    wait_async0();       // this wave's fill of As/Bs[buf] is complete
    __syncthreads();     // everyone's fill complete; prev compute done
    if (kc + 64 < K) {   // prefetch next chunk into the spare buffer
      async_cp64B(la[buf ^ 1], gA + kc + 64);
      async_cp64B(lb[buf ^ 1], gB + kc + 64);
    }
    const __bf16* Ab = As[buf];
    const __bf16* Bb = Bs[buf];
#pragma unroll
    for (int ks = 0; ks < 64; ks += 32) {
      v16bf af[2], bfr[4];
#pragma unroll
      for (int mi = 0; mi < 2; ++mi)
        af[mi] = frag_ld(Ab + ks, wm * 32 + mi * 16 + ln, 72, lane);
#pragma unroll
      for (int ni = 0; ni < 4; ++ni)
        bfr[ni] = frag_ld(Bb + ks, wn * 64 + ni * 16 + ln, 72, lane);
#pragma unroll
      for (int mi = 0; mi < 2; ++mi)
#pragma unroll
        for (int ni = 0; ni < 4; ++ni)
          acc[mi][ni] = wmma_bf16(af[mi], bfr[ni], acc[mi][ni]);
    }
  }

  float al = alpha[0];
  float scale = __powf(__fsqrt_rn((float)outF) / log1pf((float)outF), al);
#pragma unroll
  for (int mi = 0; mi < 2; ++mi) {
#pragma unroll
    for (int ni = 0; ni < 4; ++ni) {
      int colg = n0 + wn * 64 + ni * 16 + ln;
      float bsq = b_sq[colg], bia = bias[colg];
#pragma unroll
      for (int j = 0; j < 8; ++j) {
        int rowg = m0 + wm * 32 + mi * 16 + hi * 8 + j;
        float dot = acc[mi][ni][j];
        float dist = a_sq[rowg] + bsq - 2.f * dot + EPSY;
        float y = dot * dot / dist * scale + bia;
        if (MODE == 1) {
          out32[(size_t)rowg * N + colg] = y;
        } else {
          int which = colg >> 10, rem = colg & 1023;
          int h = rem >> 6, d = rem & 63;
          int bb = rowg >> 11, tp = rowg & 2047;
          size_t bh = (size_t)(bb * H_ + h);
          if (which < 2) {  // q or k: apply RoPE table factor
            int jj = d & 31;
            float ang = (float)tp * __powf(10000.f, -(float)(2 * jj) * (1.f / 64.f));
            y *= (d < 32) ? __cosf(ang) : __sinf(ang);
            __bf16* dst = (which == 0) ? qh : kh;
            dst[(bh * T_ + tp) * D_ + d] = (__bf16)y;
          } else {          // v: store transposed [B,H,D,T] for the PV wmma
            vT[(bh * D_ + d) * T_ + tp] = (__bf16)y;
          }
        }
      }
    }
  }
}

// ---- causal flash attention (64 q-rows per block, 4 waves) ----------------
// K/V tiles double-buffered + async-staged; next key block prefetch overlaps
// S-WMMA, softmax and P*V of the current block.
__global__ __launch_bounds__(128) void k_flash_attn(
    const __bf16* __restrict__ qh, const __bf16* __restrict__ kh,
    const __bf16* __restrict__ vT, __bf16* __restrict__ obf) {
  __shared__ __bf16 Qs[64 * 72];
  __shared__ __bf16 Ks[2][64 * 72];
  __shared__ __bf16 Vts[2][64 * 72];
  __shared__ __bf16 Ps[64 * 72];
  __shared__ float Ss[64 * 65];
  __shared__ float m_l[64], l_l[64], c_l[64];
  int t = threadIdx.x, lane = t & 31, wave = t >> 5;
  int wm = wave >> 1, wn = wave & 1, hi = (lane >> 4) & 1, ln = lane & 15;
  int r0 = blockIdx.x * 64, h = blockIdx.y, b = blockIdx.z;
  size_t bh = (size_t)(b * H_ + h);
  const __bf16* qg = qh + bh * T_ * D_;
  const __bf16* kg = kh + bh * T_ * D_;
  const __bf16* vg = vT + bh * D_ * T_;
  int srow = t >> 1, spart = t & 1;

  unsigned lq = lds_off(&Qs[srow * 72 + spart * 32]);
  unsigned lk[2], lv[2];
  lk[0] = lds_off(&Ks[0][srow * 72 + spart * 32]);
  lk[1] = lds_off(&Ks[1][srow * 72 + spart * 32]);
  lv[0] = lds_off(&Vts[0][srow * 72 + spart * 32]);
  lv[1] = lds_off(&Vts[1][srow * 72 + spart * 32]);
  const __bf16* gkb = kg + (size_t)srow * D_ + spart * 32;   // + kb*D_
  const __bf16* gvb = vg + (size_t)srow * T_ + spart * 32;   // + kb

  // async-stage Q tile and the first K/V block (buffer 0)
  async_cp64B(lq, qg + (size_t)(r0 + srow) * D_ + spart * 32);
  async_cp64B(lk[0], gkb);
  async_cp64B(lv[0], gvb);

  if (t < 64) { m_l[t] = -3.0e38f; l_l[t] = 0.f; }
  v8f o[2][2];
#pragma unroll
  for (int mi = 0; mi < 2; ++mi)
#pragma unroll
    for (int ni = 0; ni < 2; ++ni) o[mi][ni] = vzero8();

  for (int kb = 0; kb <= r0; kb += 64) {
    int buf = (kb >> 6) & 1;
    wait_async0();
    __syncthreads();
    if (kb + 64 <= r0) {  // prefetch next key block into spare buffer
      async_cp64B(lk[buf ^ 1], gkb + (size_t)(kb + 64) * D_);
      async_cp64B(lv[buf ^ 1], gvb + (kb + 64));
    }

    // S = (Q K^T) / sqrt(D), causal-masked; each wave 32x32 of 64x64
    v8f s[2][2];
#pragma unroll
    for (int mi = 0; mi < 2; ++mi)
#pragma unroll
      for (int ni = 0; ni < 2; ++ni) s[mi][ni] = vzero8();
#pragma unroll
    for (int ks = 0; ks < 64; ks += 32) {
      v16bf aq[2], bk[2];
#pragma unroll
      for (int mi = 0; mi < 2; ++mi)
        aq[mi] = frag_ld(Qs + ks, wm * 32 + mi * 16 + ln, 72, lane);
#pragma unroll
      for (int ni = 0; ni < 2; ++ni)
        bk[ni] = frag_ld(Ks[buf] + ks, wn * 32 + ni * 16 + ln, 72, lane);
#pragma unroll
      for (int mi = 0; mi < 2; ++mi)
#pragma unroll
        for (int ni = 0; ni < 2; ++ni)
          s[mi][ni] = wmma_bf16(aq[mi], bk[ni], s[mi][ni]);
    }
#pragma unroll
    for (int mi = 0; mi < 2; ++mi)
#pragma unroll
      for (int ni = 0; ni < 2; ++ni) {
        int col = wn * 32 + ni * 16 + ln;
#pragma unroll
        for (int j = 0; j < 8; ++j) {
          int rowm = wm * 32 + mi * 16 + hi * 8 + j;
          float sv = s[mi][ni][j] * 0.125f;
          if (kb + col > r0 + rowm) sv = -3.0e38f;  // causal
          Ss[rowm * 65 + col] = sv;
        }
      }
    __syncthreads();

    // online softmax, one thread per row
    if (t < 64) {
      float mi0 = m_l[t], li = l_l[t];
      float bm = -3.0e38f;
      for (int c2 = 0; c2 < 64; ++c2) bm = fmaxf(bm, Ss[t * 65 + c2]);
      float nm = fmaxf(mi0, bm);
      float corr = __expf(mi0 - nm);
      float sum = 0.f;
      for (int c2 = 0; c2 < 64; ++c2) {
        float p = __expf(Ss[t * 65 + c2] - nm);
        sum += p;
        Ps[t * 72 + c2] = (__bf16)p;
      }
      l_l[t] = li * corr + sum;
      m_l[t] = nm;
      c_l[t] = corr;
    }
    __syncthreads();

    // rescale running O, then O += P * V
#pragma unroll
    for (int mi = 0; mi < 2; ++mi)
#pragma unroll
      for (int ni = 0; ni < 2; ++ni)
#pragma unroll
        for (int j = 0; j < 8; ++j)
          o[mi][ni][j] *= c_l[wm * 32 + mi * 16 + hi * 8 + j];
#pragma unroll
    for (int ks = 0; ks < 64; ks += 32) {
      v16bf ap[2], bv[2];
#pragma unroll
      for (int mi = 0; mi < 2; ++mi)
        ap[mi] = frag_ld(Ps + ks, wm * 32 + mi * 16 + ln, 72, lane);
#pragma unroll
      for (int ni = 0; ni < 2; ++ni)
        bv[ni] = frag_ld(Vts[buf] + ks, wn * 32 + ni * 16 + ln, 72, lane);
#pragma unroll
      for (int mi = 0; mi < 2; ++mi)
#pragma unroll
        for (int ni = 0; ni < 2; ++ni)
          o[mi][ni] = wmma_bf16(ap[mi], bv[ni], o[mi][ni]);
    }
  }

  // epilogue: O /= l, gather heads into [B*T, C] bf16
#pragma unroll
  for (int mi = 0; mi < 2; ++mi)
#pragma unroll
    for (int ni = 0; ni < 2; ++ni) {
      int dv = wn * 32 + ni * 16 + ln;
#pragma unroll
      for (int j = 0; j < 8; ++j) {
        int rowm = wm * 32 + mi * 16 + hi * 8 + j;
        float y = o[mi][ni][j] / l_l[rowm];
        obf[((size_t)(b * T_ + r0 + rowm)) * C_ + h * 64 + dv] = (__bf16)y;
      }
    }
}

// ---------------------------------------------------------------------------
extern "C" void kernel_launch(void* const* d_in, const int* in_sizes, int n_in,
                              void* d_out, int out_size, void* d_ws, size_t ws_size,
                              hipStream_t stream) {
  (void)in_sizes; (void)n_in; (void)out_size; (void)ws_size;
  const float* x      = (const float*)d_in[0];
  // d_in[1] = mask (bool tril) -> causality is hard-coded
  const float* w_qkv  = (const float*)d_in[2];
  const float* b_qkv  = (const float*)d_in[3];
  const float* a_qkv  = (const float*)d_in[4];
  const float* w_proj = (const float*)d_in[5];
  const float* b_proj = (const float*)d_in[6];
  const float* a_proj = (const float*)d_in[7];

  char* wsp = (char*)d_ws;
  auto alloc = [&](size_t bytes) {
    void* p = (void*)wsp;
    wsp += (bytes + 255) & ~(size_t)255;
    return p;
  };
  __bf16* xbf  = (__bf16*)alloc((size_t)M_ * C_ * 2);
  float*  xsq  = (float*) alloc((size_t)M_ * 4);
  __bf16* wqT  = (__bf16*)alloc((size_t)NQKV_ * C_ * 2);
  float*  wqsq = (float*) alloc((size_t)NQKV_ * 4);
  __bf16* wpT  = (__bf16*)alloc((size_t)C_ * C_ * 2);
  float*  wpsq = (float*) alloc((size_t)C_ * 4);
  __bf16* qh   = (__bf16*)alloc((size_t)B_ * H_ * T_ * D_ * 2);
  __bf16* khb  = (__bf16*)alloc((size_t)B_ * H_ * T_ * D_ * 2);
  __bf16* vTb  = (__bf16*)alloc((size_t)B_ * H_ * T_ * D_ * 2);
  __bf16* obf  = (__bf16*)alloc((size_t)M_ * C_ * 2);
  float*  osq  = (float*) alloc((size_t)M_ * 4);

  k_convert_rowsq<<<M_, 256, 0, stream>>>(x, xbf, xsq);
  k_transpose_bf16<<<dim3(NQKV_ / 32, C_ / 32), 256, 0, stream>>>(w_qkv, wqT, C_, NQKV_);
  k_transpose_bf16<<<dim3(C_ / 32, C_ / 32), 256, 0, stream>>>(w_proj, wpT, C_, C_);
  k_rowsq_bf16<<<NQKV_, 256, 0, stream>>>(wqT, wqsq);
  k_rowsq_bf16<<<C_, 256, 0, stream>>>(wpT, wpsq);

  k_yat_gemm<0><<<dim3(NQKV_ / 128, M_ / 128), 256, 0, stream>>>(
      xbf, wqT, xsq, wqsq, b_qkv, a_qkv, NQKV_, NQKV_, nullptr, qh, khb, vTb);

  k_flash_attn<<<dim3(T_ / 64, H_, B_), 128, 0, stream>>>(qh, khb, vTb, obf);

  k_rowsq_bf16<<<M_, 256, 0, stream>>>(obf, osq);

  k_yat_gemm<1><<<dim3(C_ / 128, M_ / 128), 256, 0, stream>>>(
      obf, wpT, osq, wpsq, b_proj, a_proj, C_, C_, (float*)d_out,
      nullptr, nullptr, nullptr);
}
